// TopKSAE_29094108463920
// MI455X (gfx1250) — compile-verified, matrix operand order checked by reading, not run on hardware
//
#include <hip/hip_runtime.h>

// Problem dims (fixed by reference)
#define B_DIM 2048
#define D_DIM 4096
#define F_DIM 32768
#define K_SEL 64

typedef __attribute__((ext_vector_type(16))) __bf16 v16bf;
typedef __attribute__((ext_vector_type(8)))  __bf16 v8bf;
typedef __attribute__((ext_vector_type(4)))  __bf16 v4bf;
typedef __attribute__((ext_vector_type(8)))  float  v8f;

// ---------------- GEMM: C[M,N] = A[M,K] * B[N,K]^T + bias[N] ----------------
// NT GEMM, fp32 inputs converted to bf16 on the global->register->LDS path,
// fp32 WMMA accumulation. Double-buffered LDS, software-pipelined global
// loads: one barrier per 64-K slab, 16 v_wmma per iteration.
#define BM 128
#define BN 128
#define BK 64
#define LDKP 72   // padded LDS pitch (144B = 36 dwords -> conflict-free b128 frag loads)

__global__ __launch_bounds__(256)
void sae_gemm_bf16_nt(const float* __restrict__ A, const float* __restrict__ Bm,
                      const float* __restrict__ bias, float* __restrict__ C,
                      int N, int Kd)
{
  __shared__ __align__(16) __bf16 lA[2][BM * LDKP];
  __shared__ __align__(16) __bf16 lB[2][BN * LDKP];

  const int tid  = threadIdx.x;
  const int lane = tid & 31;
  const int wave = tid >> 5;        // 0..7
  const int wm   = wave & 3;        // 4 waves along M
  const int wn   = wave >> 2;       // 2 waves along N
  const int mBase = wm * 32;        // wave tile 32x64
  const int nBase = wn * 64;

  const int bm0 = blockIdx.y * BM;
  const int bn0 = blockIdx.x * BN;

  // global<->LDS mapping: 64 floats/row = 16 float4; 256 thr = 16 rows/batch
  const int lc = tid & 15;          // float4 index within a row
  const int lr = tid >> 4;          // 0..15 row within a batch

  v8f acc[2][4];
#pragma unroll
  for (int mi = 0; mi < 2; ++mi)
#pragma unroll
    for (int ni = 0; ni < 4; ++ni)
      acc[mi][ni] = (v8f){0.f,0.f,0.f,0.f,0.f,0.f,0.f,0.f};

  // 16-bit A/B fragment lane layout (ISA 7.12.2):
  // lanes 0-15: row = lane,    K in {0..7, 16..23}
  // lanes16-31: row = lane-16, K in {8..15, 24..31}
  const int hk = (lane < 16) ? 0 : 8;
  const int fr = lane & 15;

  v4bf sa[8], sb[8];                // staged (pre-converted) next tiles

  auto loadTiles = [&](int k0) {
#pragma unroll
    for (int r = 0; r < 8; ++r) {
      const int row = lr + r * 16;
      const float4 va = *(const float4*)&A [(size_t)(bm0 + row) * Kd + k0 + lc * 4];
      const float4 vb = *(const float4*)&Bm[(size_t)(bn0 + row) * Kd + k0 + lc * 4];
      v4bf ta, tb;
      ta[0] = (__bf16)va.x; ta[1] = (__bf16)va.y; ta[2] = (__bf16)va.z; ta[3] = (__bf16)va.w;
      tb[0] = (__bf16)vb.x; tb[1] = (__bf16)vb.y; tb[2] = (__bf16)vb.z; tb[3] = (__bf16)vb.w;
      sa[r] = ta; sb[r] = tb;
    }
  };
  auto storeTiles = [&](int buf) {
#pragma unroll
    for (int r = 0; r < 8; ++r) {
      const int row = lr + r * 16;
      *(v4bf*)&lA[buf][row * LDKP + lc * 4] = sa[r];
      *(v4bf*)&lB[buf][row * LDKP + lc * 4] = sb[r];
    }
  };

  loadTiles(0);
  storeTiles(0);
  __syncthreads();

  int buf = 0;
  for (int k0 = 0; k0 < Kd; k0 += BK) {
    const bool more = (k0 + BK) < Kd;
    if (more) loadTiles(k0 + BK);       // overlap global fetch with WMMA below

#pragma unroll
    for (int ks = 0; ks < BK; ks += 32) {
      v16bf af[2], bfr[4];
#pragma unroll
      for (int mi = 0; mi < 2; ++mi) {
        const int r = mBase + mi * 16 + fr;
        union { v16bf v; v8bf h[2]; } u;
        u.h[0] = *(const v8bf*)&lA[buf][r * LDKP + ks + hk];
        u.h[1] = *(const v8bf*)&lA[buf][r * LDKP + ks + hk + 16];
        af[mi] = u.v;
      }
#pragma unroll
      for (int ni = 0; ni < 4; ++ni) {
        const int r = nBase + ni * 16 + fr;
        union { v16bf v; v8bf h[2]; } u;
        u.h[0] = *(const v8bf*)&lB[buf][r * LDKP + ks + hk];
        u.h[1] = *(const v8bf*)&lB[buf][r * LDKP + ks + hk + 16];
        bfr[ni] = u.v;
      }
#pragma unroll
      for (int mi = 0; mi < 2; ++mi)
#pragma unroll
        for (int ni = 0; ni < 4; ++ni)
          acc[mi][ni] = __builtin_amdgcn_wmma_f32_16x16x32_bf16(
              false, af[mi], false, bfr[ni], (short)0, acc[mi][ni], false, false);
    }

    if (more) storeTiles(buf ^ 1);      // fill ping-pong buffer
    __syncthreads();                    // single barrier per 64-K slab
    buf ^= 1;
  }

  // Epilogue: C/D layout — VGPR e: M = e + (lane<16 ? 0 : 8), N = lane&15
#pragma unroll
  for (int ni = 0; ni < 4; ++ni) {
    const int col = bn0 + nBase + ni * 16 + fr;
    const float bv = bias[col];
#pragma unroll
    for (int mi = 0; mi < 2; ++mi) {
      const int rbase = bm0 + mBase + mi * 16 + ((lane < 16) ? 0 : 8);
#pragma unroll
      for (int e = 0; e < 8; ++e)
        C[(size_t)(rbase + e) * N + col] = acc[mi][ni][e] + bv;
    }
  }
}

// ---------------- Top-K select (per row), relu, scatter, indices ----------------
// One block per row. Row cached in LDS (128KB; CDNA5 allows up to 320KB/WG).
// 4-pass 8-bit radix select on monotonic uint keys, then deterministic
// ballot-prefix compaction (no ordering atomics -> bit-stable across replays).
#define TPB 256

__device__ __forceinline__ unsigned int sae_key(float f) {
  unsigned int u = __float_as_uint(f);
  return (u & 0x80000000u) ? ~u : (u | 0x80000000u);
}

__global__ __launch_bounds__(TPB)
void sae_topk_scatter(const float* __restrict__ h, float* __restrict__ hs,
                      int* __restrict__ idx_out)
{
  __shared__ __align__(16) float row[F_DIM];    // 128 KB
  __shared__ unsigned int hist[256];
  __shared__ unsigned int ctrl[2];              // [0]=prefix, [1]=remaining
  __shared__ unsigned int wG[8], wE[8];

  const int b   = blockIdx.x;
  const int tid = threadIdx.x;
  const int lane = tid & 31;
  const int wave = tid >> 5;
  const float* hrow = h + (size_t)b * F_DIM;

  for (int i = tid; i < F_DIM / 4; i += TPB)
    *(float4*)&row[i * 4] = *(const float4*)&hrow[i * 4];
  if (tid == 0) { ctrl[0] = 0u; ctrl[1] = K_SEL; }
  __syncthreads();

  // radix select: find key of the K_SEL-th largest element
  for (int p = 0; p < 4; ++p) {
    const int shift = 24 - 8 * p;
    if (tid < 256) hist[tid] = 0u;
    __syncthreads();
    const unsigned int prefix = ctrl[0];
    for (int i = tid; i < F_DIM; i += TPB) {
      const unsigned int key = sae_key(row[i]);
      const bool match = (p == 0) || ((key >> (shift + 8)) == prefix);
      if (match) atomicAdd(&hist[(key >> shift) & 255u], 1u);
    }
    __syncthreads();
    if (tid == 0) {
      const unsigned int rem = ctrl[1];
      unsigned int cum = 0; int d = 255;
      for (; d > 0; --d) {
        const unsigned int c = hist[d];
        if (cum + c >= rem) break;
        cum += c;
      }
      ctrl[1] = rem - cum;
      ctrl[0] = (prefix << 8) | (unsigned int)d;
    }
    __syncthreads();
  }
  const unsigned int T = ctrl[0];   // threshold key (K-th largest)
  const unsigned int E = ctrl[1];   // #elements equal to T to take (lowest index first)

  float* hsrow = hs + (size_t)b * F_DIM;
  int*   irow  = idx_out + b * K_SEL;
  unsigned int runG = 0, runE = 0;  // uniform running totals

  for (int c = 0; c < F_DIM / TPB; ++c) {
    const int i = c * TPB + tid;
    const float v = row[i];
    const unsigned int key = sae_key(v);
    const bool g = key > T;
    const bool e = key == T;
    const unsigned int mg = (unsigned int)__ballot(g);
    const unsigned int me = (unsigned int)__ballot(e);
    if (lane == 0) { wG[wave] = __popc(mg); wE[wave] = __popc(me); }
    __syncthreads();
    unsigned int pg = 0, pe = 0, totG = 0, totE = 0;
    for (int w = 0; w < 8; ++w) {
      if (w < wave) { pg += wG[w]; pe += wE[w]; }
      totG += wG[w]; totE += wE[w];
    }
    const unsigned int low = (lane == 0) ? 0u : (0xFFFFFFFFu >> (32 - lane));
    const unsigned int gPre = runG + pg + __popc(mg & low);
    const unsigned int ePre = runE + pe + __popc(me & low);
    const bool sel = g || (e && ePre < E);
    if (sel) {
      const unsigned int pos = gPre + (ePre < E ? ePre : E);
      if (pos < K_SEL) irow[pos] = i;   // index order (set matches reference)
    }
    hsrow[i] = sel ? (v > 0.f ? v : 0.f) : 0.f;   // relu after select
    runG += totG; runE += totE;
    __syncthreads();
  }
}

// ---------------- launch ----------------
extern "C" void kernel_launch(void* const* d_in, const int* in_sizes, int n_in,
                              void* d_out, int out_size, void* d_ws, size_t ws_size,
                              hipStream_t stream) {
  const float* x     = (const float*)d_in[0];   // (B, D)
  const float* W_enc = (const float*)d_in[1];   // (F, D)
  const float* b_enc = (const float*)d_in[2];   // (F,)
  const float* W_dec = (const float*)d_in[3];   // (D, F)
  const float* b_dec = (const float*)d_in[4];   // (D,)

  float* out      = (float*)d_out;
  float* x_hat    = out;                                          // B*D
  float* h_sparse = out + (size_t)B_DIM * D_DIM;                  // B*F
  int*   idx      = (int*)(out + (size_t)B_DIM * D_DIM + (size_t)B_DIM * F_DIM); // B*K

  float* h = (float*)d_ws;  // dense pre-activations, B*F*4 = 256 MB scratch

  // 1) encode: h = x @ W_enc^T + b_enc   (M=B, N=F, K=D)
  sae_gemm_bf16_nt<<<dim3(F_DIM / BN, B_DIM / BM), 256, 0, stream>>>(
      x, W_enc, b_enc, h, F_DIM, D_DIM);

  // 2) per-row top-64 -> h_sparse (relu'd, scattered) + indices
  sae_topk_scatter<<<B_DIM, TPB, 0, stream>>>(h, h_sparse, idx);

  // 3) decode: x_hat = h_sparse @ W_dec^T + b_dec   (M=B, N=D, K=F)
  sae_gemm_bf16_nt<<<dim3(D_DIM / BN, B_DIM / BM), 256, 0, stream>>>(
      h_sparse, W_dec, b_dec, x_hat, D_DIM, F_DIM);
}
